// HypothesisDecoder_17136919511625
// MI455X (gfx1250) — matrix operand order repaired and checked by reading, not compiled, not run on hardware
//
#include <hip/hip_runtime.h>
#include <hip/hip_bf16.h>
#include <math.h>

typedef __attribute__((ext_vector_type(16))) _Float16 v16h;
typedef __attribute__((ext_vector_type(8)))  _Float16 v8h;
typedef __attribute__((ext_vector_type(8)))  float    v8f;

#define N_PTS 8192
#define H_HYP 32
#define C0 128
#define C1 128
#define CF 64
#define IN_DIM 320   // concat order: f1(128), f0(128), pts_feat(64)
#define HID 256
#define D0 24
#define D1 48
#define EPS 1e-5f

#define PPB 2        // points per block (amortizes L2 weight traffic)
// LDS row strides padded for bank-conflict-free 16-lane row sweeps
// (64 banks x 4B): 328 halves = 164 dw (164%64=36), 264 halves = 132 dw (132%64=4)
#define LDA0 328
#define LDA  264
#define PT0 (34 * LDA0)   // per-point stride in sX0 (halves)
#define PT1 (34 * LDA)    // per-point stride in sX1/sX2

// ---- workspace layout (halves then floats) ----
#define T0_HALVES (16*30*512)   // tiled w0 (Ktot=960 -> 30 chunks)
#define T1_HALVES (16*24*512)   // tiled w1 (Ktot=768 -> 24 chunks)
#define T2_HALVES (16*24*512)   // tiled w2
#define PREP_TOTAL (T0_HALVES + T1_HALVES + T2_HALVES)

// =====================================================================
// Prep: tile conv weights into WMMA-B-fragment-ready f16 layout
// (lane = N col + 16*khalf, 16 K halves per lane in table order), fold BN into scale/bias.
// =====================================================================
__device__ __forceinline__ void tile_weight(_Float16* dst, const float* w,
                                            int Cin, int nch, int idx) {
  int nt  = idx / (nch * 512);
  int rem = idx % (nch * 512);
  int c   = rem / 512;
  int r2  = rem % 512;
  int ln  = r2 / 16, s = r2 % 16;
  int nn  = ln % 16;
  int kt  = (s < 8) ? 8 * (ln / 16) + s : 16 + 8 * (ln / 16) + (s - 8);
  int k   = (32 * c) / Cin;          // constant within chunk (Cin % 32 == 0)
  int i   = (32 * c) % Cin + kt;
  int o   = nt * 16 + nn;
  dst[idx] = (_Float16)w[((size_t)o * Cin + i) * 3 + k];
}

__global__ void prep_kernel(const float* __restrict__ w0,
                            const float* __restrict__ w1,
                            const float* __restrict__ w2,
                            const float* __restrict__ g0, const float* __restrict__ b0,
                            const float* __restrict__ m0, const float* __restrict__ v0,
                            const float* __restrict__ g1, const float* __restrict__ b1,
                            const float* __restrict__ m1, const float* __restrict__ v1,
                            const float* __restrict__ g2, const float* __restrict__ b2,
                            const float* __restrict__ m2, const float* __restrict__ v2,
                            _Float16* __restrict__ t0, _Float16* __restrict__ t1,
                            _Float16* __restrict__ t2, float* __restrict__ SB) {
  int gid = blockIdx.x * blockDim.x + threadIdx.x;
  if (gid < T0_HALVES) {
    tile_weight(t0, w0, IN_DIM, 30, gid);
  } else if (gid < T0_HALVES + T1_HALVES) {
    tile_weight(t1, w1, HID, 24, gid - T0_HALVES);
  } else if (gid < PREP_TOTAL) {
    tile_weight(t2, w2, HID, 24, gid - T0_HALVES - T1_HALVES);
  }
  if (gid < 3 * HID) {
    int l = gid / HID, o = gid % HID;
    const float *g, *b, *m, *v;
    if (l == 0)      { g = g0; b = b0; m = m0; v = v0; }
    else if (l == 1) { g = g1; b = b1; m = m1; v = v1; }
    else             { g = g2; b = b2; m = m2; v = v2; }
    float s = g[o] * rsqrtf(v[o] + EPS);
    SB[l * 512 + o]       = s;
    SB[l * 512 + 256 + o] = b[o] - m[o] * s;
  }
}

// =====================================================================
// A-fragment from padded LDS activations (row r == hypothesis h = r-1;
// rows 0 and 33 are zero pad). ISA 16-bit A 16x32 layout:
// lane<16: M=lane, K in {0..7,16..23}; lane>=16: M=lane-16, K in {8..15,24..31}
// =====================================================================
__device__ __forceinline__ v16h load_a(const _Float16* X, int ld, int mt,
                                       int k, int i0, int lane) {
  int m  = lane & 15;
  int kh = lane >> 4;
  const _Float16* p = X + (mt * 16 + m + k) * ld + i0 + kh * 8;
  v8h lo = *(const v8h*)p;
  v8h hi = *(const v8h*)(p + 16);
  v16h a;
#pragma unroll
  for (int j = 0; j < 8; ++j) { a[j] = lo[j]; a[j + 8] = hi[j]; }
  return a;
}

__device__ __forceinline__ v16h load_b(const _Float16* Wt, int nt, int nch,
                                       int c, int lane) {
  return *(const v16h*)(Wt + (((size_t)(nt * nch + c)) * 32 + lane) * 16);
}

// One conv layer as WMMA GEMM over PPB points: A [32 x 3*Cin] per point
// (im2col via shifted LDS rows), B [3*Cin x 256] pre-tiled in global.
// Each wave owns 2 N-tiles x (PPB*2) M-tiles -> each B fragment feeds 4 WMMAs.
__device__ __forceinline__ void layer_gemm(const _Float16* Xin, int ldin, int pstrIn,
                                           _Float16* Xout, int ldout, int pstrOut,
                                           const _Float16* Wt, int Cin,
                                           const float* scale, const float* bias,
                                           int tid) {
  const int nch  = (3 * Cin) >> 5;
  const int wave = tid >> 5, lane = tid & 31;
  const int nt0 = wave * 2, nt1 = wave * 2 + 1;
  v8f acc[PPB * 2][2] = {};   // [p*2+mt][ntile]
  for (int c = 0; c < nch; ++c) {
    int k  = (32 * c) / Cin;
    int i0 = (32 * c) % Cin;
    v16h b0 = load_b(Wt, nt0, nch, c, lane);
    v16h b1 = load_b(Wt, nt1, nch, c, lane);
#pragma unroll
    for (int u = 0; u < PPB * 2; ++u) {
      int p = u >> 1, mt = u & 1;
      v16h a = load_a(Xin + p * pstrIn, ldin, mt, k, i0, lane);
      acc[u][0] = __builtin_amdgcn_wmma_f32_16x16x32_f16(false, a, false, b0, (short)0, acc[u][0], false, false);
      acc[u][1] = __builtin_amdgcn_wmma_f32_16x16x32_f16(false, a, false, b1, (short)0, acc[u][1], false, false);
    }
  }
  // C/D layout: lane -> N = lane&15, M = r + 8*(lane>>4)
  int ncol = lane & 15, rb = (lane >> 4) * 8;
  int o0 = nt0 * 16 + ncol, o1 = nt1 * 16 + ncol;
  float s0 = scale[o0], t0v = bias[o0];
  float s1 = scale[o1], t1v = bias[o1];
#pragma unroll
  for (int u = 0; u < PPB * 2; ++u) {
    int p = u >> 1, mt = u & 1;
    _Float16* Xo = Xout + p * pstrOut;
#pragma unroll
    for (int r = 0; r < 8; ++r) {
      int h = mt * 16 + rb + r;
      Xo[(h + 1) * ldout + o0] = (_Float16)fmaxf(acc[u][0][r] * s0 + t0v, 0.f);
      Xo[(h + 1) * ldout + o1] = (_Float16)fmaxf(acc[u][1][r] * s1 + t1v, 0.f);
    }
  }
}

// =====================================================================
// Fused kernel: trilinear gather -> 3 WMMA conv layers -> head + softmax.
// One block (256 threads = 8 waves) per PPB points; activations live in LDS.
// =====================================================================
__global__ void __launch_bounds__(256)
decoder_kernel(const float* __restrict__ pts,
               const float* __restrict__ pts_feat,
               const int*   __restrict__ pts_batch,
               const float* __restrict__ vol0,
               const float* __restrict__ vol1,
               const _Float16* __restrict__ t0,
               const _Float16* __restrict__ t1,
               const _Float16* __restrict__ t2,
               const float* __restrict__ SB,
               const float* __restrict__ w3,
               const float* __restrict__ b3,
               float* __restrict__ out) {
  __shared__ __align__(16) _Float16 sX0[PPB * PT0];
  __shared__ __align__(16) _Float16 sX1[PPB * PT1];
  __shared__ __align__(16) _Float16 sX2[PPB * PT1];
  __shared__ float sW[PPB][2][H_HYP][8];   // per-corner weight*valid
  __shared__ int   sO[PPB][2][H_HYP][8];   // per-corner clamped flat offset
  __shared__ float sPart[PPB][H_HYP][8];
  __shared__ int   sBat[PPB];

  const int n0  = blockIdx.x * PPB;
  const int tid = threadIdx.x;

  if (tid < PPB) sBat[tid] = pts_batch[n0 + tid];

  // zero the conv padding rows (h = -1 and h = 32) for every point
  for (int j = tid; j < PPB * IN_DIM; j += 256) {
    int p = j / IN_DIM, c = j % IN_DIM;
    sX0[p * PT0 + c] = (_Float16)0.f;
    sX0[p * PT0 + 33 * LDA0 + c] = (_Float16)0.f;
  }
  for (int j = tid; j < PPB * HID; j += 256) {
    int p = j / HID, c = j % HID;
    sX1[p * PT1 + c] = (_Float16)0.f; sX1[p * PT1 + 33 * LDA + c] = (_Float16)0.f;
    sX2[p * PT1 + c] = (_Float16)0.f; sX2[p * PT1 + 33 * LDA + c] = (_Float16)0.f;
  }

  // ---- corner weights / offsets: 128 threads = (point, vol, hypothesis) ----
  if (tid < PPB * 64) {
    int p = tid >> 6, vi = (tid >> 5) & 1, h = tid & 31;
    int D = vi ? D1 : D0;
    const float* q = pts + ((size_t)(n0 + p) * H_HYP + h) * 3;
    float cx = q[0] * (D - 1), cy = q[1] * (D - 1), cz = q[2] * (D - 1);
    float fx = floorf(cx), fy = floorf(cy), fz = floorf(cz);
    float rx = cx - fx, ry = cy - fy, rz = cz - fz;
    int ix = (int)fx, iy = (int)fy, iz = (int)fz;
#pragma unroll
    for (int corner = 0; corner < 8; ++corner) {
      int dx = (corner >> 2) & 1, dy = (corner >> 1) & 1, dz = corner & 1;
      int jx = ix + dx, jy = iy + dy, jz = iz + dz;
      bool valid = (jx >= 0) && (jx < D) && (jy >= 0) && (jy < D) && (jz >= 0) && (jz < D);
      int cxc = min(max(jx, 0), D - 1);
      int cyc = min(max(jy, 0), D - 1);
      int czc = min(max(jz, 0), D - 1);
      float w = (dx ? rx : 1.f - rx) * (dy ? ry : 1.f - ry) * (dz ? rz : 1.f - rz);
      sW[p][vi][h][corner] = valid ? w : 0.f;
      sO[p][vi][h][corner] = (cxc * D + cyc) * D + czc;
    }
  }
  __syncthreads();

  // ---- gather: f1 (vol1) -> cols [0,128), f0 (vol0) -> [128,256) ----
  for (int it = tid; it < PPB * H_HYP * C1; it += 256) {          // vol1
    int p = it >> 12, w = it & 4095;
    int h = w & 31, c = w >> 5;
    const float* base = vol1 + ((size_t)sBat[p] * C1 + c) * (D1 * D1 * D1);
    float acc = 0.f;
#pragma unroll
    for (int corner = 0; corner < 8; ++corner)
      acc += sW[p][1][h][corner] * base[sO[p][1][h][corner]];
    sX0[p * PT0 + (h + 1) * LDA0 + c] = (_Float16)acc;
  }
  for (int it = tid; it < PPB * H_HYP * C0; it += 256) {          // vol0
    int p = it >> 12, w = it & 4095;
    int h = w & 31, c = w >> 5;
    const float* base = vol0 + ((size_t)sBat[p] * C0 + c) * (D0 * D0 * D0);
    float acc = 0.f;
#pragma unroll
    for (int corner = 0; corner < 8; ++corner)
      acc += sW[p][0][h][corner] * base[sO[p][0][h][corner]];
    sX0[p * PT0 + (h + 1) * LDA0 + 128 + c] = (_Float16)acc;
  }
  for (int it = tid; it < PPB * H_HYP * CF; it += 256) {          // pts_feat
    int p = it >> 11, w = it & 2047;
    int c = w & 63, h = w >> 6;
    sX0[p * PT0 + (h + 1) * LDA0 + 256 + c] =
        (_Float16)pts_feat[((size_t)(n0 + p) * H_HYP + h) * CF + c];
  }
  __syncthreads();

  // ---- conv layers as WMMA GEMMs ----
  layer_gemm(sX0, LDA0, PT0, sX1, LDA, PT1, t0, IN_DIM, SB + 0 * 512, SB + 0 * 512 + 256, tid);
  __syncthreads();
  layer_gemm(sX1, LDA, PT1, sX2, LDA, PT1, t1, HID, SB + 1 * 512, SB + 1 * 512 + 256, tid);
  __syncthreads();
  layer_gemm(sX2, LDA, PT1, sX1, LDA, PT1, t2, HID, SB + 2 * 512, SB + 2 * 512 + 256, tid);
  __syncthreads();

  // ---- head: 256->1 conv (k=3) partials, 8 threads per (point, hypothesis) ----
  for (int it = tid; it < PPB * 256; it += 256) {
    int p = it >> 8, r = it & 255;
    int h = r >> 3, j = r & 7;
    float part = 0.f;
#pragma unroll
    for (int k = 0; k < 3; ++k) {
      const _Float16* row = sX1 + p * PT1 + (h + k) * LDA + j * 32;
      const float* wv = w3 + (j * 32) * 3 + k;
#pragma unroll
      for (int ii = 0; ii < 32; ++ii)
        part += wv[ii * 3] * (float)row[ii];
    }
    sPart[p][h][j] = part;
  }
  __syncthreads();

  // ---- softmax over 32 hypotheses: wave p handles point p ----
  if (tid < PPB * 32) {
    int p = tid >> 5, h = tid & 31;
    float logit = b3[0];
#pragma unroll
    for (int j = 0; j < 8; ++j) logit += sPart[p][h][j];
    float mx = logit;
#pragma unroll
    for (int s = 16; s > 0; s >>= 1) mx = fmaxf(mx, __shfl_xor(mx, s, 32));
    float e = __expf(logit - mx);
    float sum = e;
#pragma unroll
    for (int s = 16; s > 0; s >>= 1) sum += __shfl_xor(sum, s, 32);
    out[(size_t)(n0 + p) * H_HYP + h] = e / sum;
  }
}

extern "C" void kernel_launch(void* const* d_in, const int* in_sizes, int n_in,
                              void* d_out, int out_size, void* d_ws, size_t ws_size,
                              hipStream_t stream) {
  const float* pts      = (const float*)d_in[0];
  const float* pts_feat = (const float*)d_in[1];
  const int*   pts_b    = (const int*)d_in[2];
  const float* vol0     = (const float*)d_in[3];
  const float* vol1     = (const float*)d_in[4];
  const float* w0       = (const float*)d_in[5];
  const float* w1       = (const float*)d_in[6];
  const float* w2       = (const float*)d_in[7];
  const float* w3       = (const float*)d_in[8];
  const float* b3       = (const float*)d_in[9];
  const float* g0 = (const float*)d_in[10], *b0 = (const float*)d_in[11];
  const float* m0 = (const float*)d_in[12], *v0 = (const float*)d_in[13];
  const float* g1 = (const float*)d_in[14], *b1 = (const float*)d_in[15];
  const float* m1 = (const float*)d_in[16], *v1 = (const float*)d_in[17];
  const float* g2 = (const float*)d_in[18], *b2 = (const float*)d_in[19];
  const float* m2 = (const float*)d_in[20], *v2 = (const float*)d_in[21];

  _Float16* t0 = (_Float16*)d_ws;
  _Float16* t1 = t0 + T0_HALVES;
  _Float16* t2 = t1 + T1_HALVES;
  float*    SB = (float*)(t2 + T2_HALVES);

  prep_kernel<<<(PREP_TOTAL + 255) / 256, 256, 0, stream>>>(
      w0, w1, w2, g0, b0, m0, v0, g1, b1, m1, v1, g2, b2, m2, v2, t0, t1, t2, SB);

  decoder_kernel<<<N_PTS / PPB, 256, 0, stream>>>(
      pts, pts_feat, pts_b, vol0, vol1, t0, t1, t2, SB, w3, b3, (float*)d_out);
}
